// GCN2_27032524161269
// MI455X (gfx1250) — matrix operand order repaired and checked by reading, not compiled, not run on hardware
//
#include <hip/hip_runtime.h>
#include <cstdint>

typedef __attribute__((ext_vector_type(16))) __bf16 v16bf;
typedef __attribute__((ext_vector_type(8)))  float  v8f;

// ---------------- degree / norm kernels ----------------

__global__ void k_fill1(float* deg, int n) {
    int i = blockIdx.x * blockDim.x + threadIdx.x;
    if (i < n) deg[i] = 1.0f;   // self loop
}

__global__ void k_deg_count(const long long* __restrict__ dst, float* deg, long long E) {
    long long e = (long long)blockIdx.x * blockDim.x + threadIdx.x;
    if (e < E) atomicAdd(&deg[(int)dst[e]], 1.0f);
}

__global__ void k_rsqrt(float* dinv, int n) {
    int i = blockIdx.x * blockDim.x + threadIdx.x;
    if (i < n) {
        float d = dinv[i];
        dinv[i] = (d > 0.0f) ? rsqrtf(d) : 0.0f;
    }
}

// ---------------- bf16 WMMA GEMM: C[M,N] = A[M,K] @ W[K,N] ----------------
// One wave per 16x16 tile; requires M % 16 == 0 (launcher guarantees it).
// A fragment: 4x global_load_b128 per K-step. B fragment: 16 strided b32
// (coalesced across the 16 consecutive columns held by lanes 0-15 / 16-31).
// Out-of-range columns are CLAMPED, never predicated: garbage flows only
// into lanes whose stores are masked once in the epilogue.

template <int RELU>
__global__ void wmma_gemm_bf16(const float* __restrict__ A,
                               const float* __restrict__ W,
                               float* __restrict__ C,
                               int M, int K, int N) {
    const int wave   = threadIdx.x >> 5;
    const int lane   = threadIdx.x & 31;
    const int half   = lane >> 4;      // 0: lanes 0-15, 1: lanes 16-31
    const int l15    = lane & 15;
    const int tile_m = blockIdx.x;
    const int tile_n = blockIdx.y * (blockDim.x >> 5) + wave;
    if (tile_n * 16 >= N) return;      // wave-uniform

    const int  row    = tile_m * 16 + l15;          // M % 16 == 0 -> in range
    const int  col    = tile_n * 16 + l15;
    const bool cvalid = (col < N);
    const int  colc   = cvalid ? col : (N - 1);     // clamp (store masked)

    const float4* __restrict__ arow4 = (const float4*)(A + (size_t)row * K);
    const float*  __restrict__ wcol  = W + colc;

    auto cv = [&](float x) -> __bf16 {
        if (RELU) x = fmaxf(x, 0.0f);
        return (__bf16)x;
    };

    v8f acc = {};
    for (int k0 = 0; k0 < K; k0 += 32) {
        const int ia = (k0 >> 2) + half * 2;   // float4 index of first chunk
        float4 q0 = arow4[ia];                 // K = k0+half*8 .. +3
        float4 q1 = arow4[ia + 1];             // .. +7
        float4 q2 = arow4[ia + 4];             // K = k0+half*8+16 .. +19
        float4 q3 = arow4[ia + 5];             // .. +23
        __builtin_prefetch((const float*)arow4 + k0 + 64, 0, 3);

        v16bf a;
        a[0]  = cv(q0.x); a[1]  = cv(q0.y); a[2]  = cv(q0.z); a[3]  = cv(q0.w);
        a[4]  = cv(q1.x); a[5]  = cv(q1.y); a[6]  = cv(q1.z); a[7]  = cv(q1.w);
        a[8]  = cv(q2.x); a[9]  = cv(q2.y); a[10] = cv(q2.z); a[11] = cv(q2.w);
        a[12] = cv(q3.x); a[13] = cv(q3.y); a[14] = cv(q3.z); a[15] = cv(q3.w);

        // B fragment: column colc, K rows half*8+0..7 and 16+half*8+0..7.
        v16bf b;
        const size_t kbN = (size_t)(k0 + half * 8) * N;
#pragma unroll
        for (int j = 0; j < 16; ++j)
            b[j] = (__bf16)wcol[kbN + (size_t)(j + (j & 8)) * N];

        acc = __builtin_amdgcn_wmma_f32_16x16x32_bf16(
            false, a, false, b, (short)0, acc, false, false);
    }

    // Epilogue: 8 plain stores down one column; single base address,
    // pointer bumped by N per row (no per-element guards or 64-bit muls).
    if (cvalid) {
        float* cp = C + (size_t)(tile_m * 16 + half * 8) * N + col;
#pragma unroll
        for (int v = 0; v < 8; ++v) {
            *cp = acc[v];
            cp += N;
        }
    }
}

// ---------------- self-loop + bias seed: out[i,f] = h[i,f]*dinv[i]^2 + b[f] ----------------
// One block per node; threads stride features (no div/mod, dinv uniform).

__global__ void k_selfbias(const float* __restrict__ h, const float* __restrict__ dinv,
                           const float* __restrict__ b, float* __restrict__ out,
                           int F) {
    const int i = blockIdx.x;
    const float di = dinv[i];
    const float di2 = di * di;
    const float* hi = h + (size_t)i * F;
    float* oi = out + (size_t)i * F;
    for (int f = threadIdx.x; f < F; f += blockDim.x)
        oi[f] = hi[f] * di2 + b[f];
}

// ---------------- edge scatter: out[dst] += h[src] * dinv[src]*dinv[dst] ----------------
// One wave per edge; lanes stride the feature dimension (coalesced L2-resident atomics).

__global__ void k_edge_agg(const long long* __restrict__ src,
                           const long long* __restrict__ dst,
                           const float* __restrict__ h,
                           const float* __restrict__ dinv,
                           float* __restrict__ out,
                           int F, long long E) {
    long long e = (long long)blockIdx.x * (blockDim.x >> 5) + (threadIdx.x >> 5);
    if (e >= E) return;
    int lane = threadIdx.x & 31;
    int s = (int)src[e];
    int d = (int)dst[e];
    float nrm = dinv[s] * dinv[d];
    const float* hs = h + (size_t)s * F;
    float* od = out + (size_t)d * F;
    for (int f = lane; f < F; f += 32)
        atomicAdd(&od[f], hs[f] * nrm);
}

// ---------------- log-softmax over 40 classes, one wave per row ----------------

__global__ void k_log_softmax(float* __restrict__ out, int n, int C) {
    int node = blockIdx.x * (blockDim.x >> 5) + (threadIdx.x >> 5);
    if (node >= n) return;
    int lane = threadIdx.x & 31;
    float* row = out + (size_t)node * C;

    float a0 = row[lane];                                   // C=40 >= 32
    bool  v1 = (lane + 32) < C;
    float a1 = v1 ? row[lane + 32] : -3.4e38f;

    float m = fmaxf(a0, a1);
#pragma unroll
    for (int off = 16; off > 0; off >>= 1)
        m = fmaxf(m, __shfl_xor(m, off, 32));

    float s = __expf(a0 - m) + (v1 ? __expf(a1 - m) : 0.0f);
#pragma unroll
    for (int off = 16; off > 0; off >>= 1)
        s += __shfl_xor(s, off, 32);

    float lse = m + __logf(s);
    row[lane] = a0 - lse;
    if (v1) row[lane + 32] = a1 - lse;
}

// ---------------- launcher ----------------

extern "C" void kernel_launch(void* const* d_in, const int* in_sizes, int n_in,
                              void* d_out, int out_size, void* d_ws, size_t ws_size,
                              hipStream_t stream) {
    const float*     x   = (const float*)d_in[0];
    const long long* ei  = (const long long*)d_in[1];   // int64 edge_index [2,E]
    const float*     W1  = (const float*)d_in[2];
    const float*     b1  = (const float*)d_in[3];
    const float*     W2  = (const float*)d_in[4];
    const float*     b2  = (const float*)d_in[5];

    const int nfeat = 512, nhid = 256, nclass = 40;
    const int n = in_sizes[0] / nfeat;                  // 100000 (divisible by 16)
    const long long E = (long long)in_sizes[1] / 2;     // 3200000
    const long long* src = ei;
    const long long* dst = ei + E;

    // workspace: dinv | h (n*nhid, reused for h2) | agg (n*nhid)
    char* ws = (char*)d_ws;
    size_t off = ((size_t)n * 4 + 255) & ~(size_t)255;
    float* dinv = (float*)ws;
    float* h    = (float*)(ws + off);
    float* agg  = (float*)(ws + off + (size_t)n * nhid * 4);
    float* out  = (float*)d_out;

    const int ntile_m = n / 16;                         // 6250 (M % 16 == 0)

    // degrees -> dinv
    k_fill1    <<<(n + 255) / 256, 256, 0, stream>>>(dinv, n);
    k_deg_count<<<(int)((E + 255) / 256), 256, 0, stream>>>(dst, dinv, E);
    k_rsqrt    <<<(n + 255) / 256, 256, 0, stream>>>(dinv, n);

    // layer 1: h = X @ W1  (16 n-tiles: 8 waves/block * gridDim.y=2)
    wmma_gemm_bf16<0><<<dim3(ntile_m, 2), dim3(256), 0, stream>>>(x, W1, h, n, nfeat, nhid);
    // agg = h*dinv^2 + b1, then edge scatter-add
    k_selfbias<<<n, 256, 0, stream>>>(h, dinv, b1, agg, nhid);
    k_edge_agg<<<(int)((E + 7) / 8), 256, 0, stream>>>(src, dst, h, dinv, agg, nhid, E);

    // layer 2: h2 = relu(agg) @ W2  (3 n-tiles: 3 waves/block), reuse h buffer
    wmma_gemm_bf16<1><<<dim3(ntile_m, 1), dim3(96), 0, stream>>>(agg, W2, h, n, nhid, nclass);
    k_selfbias<<<n, 64, 0, stream>>>(h, dinv, b2, out, nclass);
    k_edge_agg<<<(int)((E + 7) / 8), 256, 0, stream>>>(src, dst, h, dinv, out, nclass, E);

    // log-softmax rows (8 waves per block)
    k_log_softmax<<<(n + 7) / 8, 256, 0, stream>>>(out, n, nclass);
}
